// Raindrop_12652973654156
// MI455X (gfx1250) — matrix operand order repaired and checked by reading, not compiled, not run on hardware
//
#include <hip/hip_runtime.h>
#include <math.h>

// ---------------- shapes ----------------
#define BB 32
#define TT 128
#define SS 64
#define HH 4
#define DD 16
#define PP 16
#define AA 12
#define TAH 12
#define AP 28          // A + P
#define DP 32          // D + P
#define LL 2
#define OUTH 16        // OUT / H

// padded LDS strides (odd bank counts -> conflict-free fragment loads)
#define HS 18          // sh_h row stride (halves)
#define HPS 34         // sh_hp row stride
#define KVS 66         // sh_kvT row stride
#define FINS 66        // sh_fin row stride
#define HCS 34         // Hc row stride

typedef __attribute__((ext_vector_type(16))) _Float16 v16h;
typedef __attribute__((ext_vector_type(8)))  float    v8f;

__device__ __forceinline__ v8f wmma_f16(v16h a, v16h b, v8f c) {
  return __builtin_amdgcn_wmma_f32_16x16x32_f16(
      false, a, false, b, (short)0, c, false, false);
}

// A fragment 16x32 f16 (M x K). lane<16: M=lane, kbase=0 ; lane>=16: M=lane-16, kbase=8
// VGPR v: K = (v>=4?16:0) + kbase + 2*(v&3)
__device__ __forceinline__ v16h load_a_frag(const _Float16* base, int row0,
                                            int stride, int lane) {
  int m  = row0 + (lane & 15);
  int kb = (lane >> 4) << 3;
  v16h a;
#pragma unroll
  for (int v = 0; v < 8; ++v) {
    int k = ((v & 4) << 2) + kb + ((v & 3) << 1);
    a[2 * v]     = base[m * stride + k];
    a[2 * v + 1] = base[m * stride + k + 1];
  }
  return a;
}

// B fragment 32x16 f16 (K x N), source row-major src[k][n]: N = lane&15,
// K per element same as A fragment: K = (v>=4?16:0) + kbase + 2*(v&3)
__device__ __forceinline__ v16h load_b_frag(const _Float16* base, int k0,
                                            int stride, int lane) {
  v16h b;
  int n  = lane & 15;
  int kb = (lane >> 4) << 3;
#pragma unroll
  for (int v = 0; v < 8; ++v) {
    int k = k0 + ((v & 4) << 2) + kb + ((v & 3) << 1);
    b[2 * v]     = base[k * stride + n];
    b[2 * v + 1] = base[(k + 1) * stride + n];
  }
  return b;
}

// ---------------- precompute kernels ----------------

__global__ void k_pe(const float* __restrict__ times, float* __restrict__ pe) {
  int idx = blockIdx.x * blockDim.x + threadIdx.x;     // B*T*8
  if (idx >= BB * TT * 8) return;
  int i = idx & 7;
  int bt = idx >> 3;                                   // b*T + t
  float div = expf(-(float)i * logf(10000.f) / 8.f);
  float ang = times[bt] * div;
  pe[bt * PP + 2 * i]     = sinf(ang);
  pe[bt * PP + 2 * i + 1] = cosf(ang);
}

__global__ void k_h0(const float* __restrict__ x, const float* __restrict__ mask,
                     const float* __restrict__ obs_w, _Float16* __restrict__ hG) {
  int idx = blockIdx.x * blockDim.x + threadIdx.x;     // B*H*T*S*D = 16.7M
  if (idx >= BB * HH * TT * SS * DD) return;
  int d  = idx & 15;
  int s  = (idx >> 4) & 63;
  int t  = (idx >> 10) & 127;
  int hh = (idx >> 17) & 3;
  int b  = idx >> 19;
  float v = x[(b * TT + t) * SS + s] * obs_w[s * 64 + hh * DD + d];
  v = fmaxf(v, 0.f) * mask[(b * TT + t) * SS + s];
  hG[idx] = (_Float16)v;
}

__global__ void k_msum(const float* __restrict__ mask, float* __restrict__ msum) {
  int idx = blockIdx.x * blockDim.x + threadIdx.x;     // B*S
  if (idx >= BB * SS) return;
  int b = idx >> 6, s = idx & 63;
  float acc = 0.f;
  for (int t = 0; t < TT; ++t) acc += mask[(b * TT + t) * SS + s];
  msum[idx] = acc;
}

__global__ void k_bidir(const float* __restrict__ bw, float* __restrict__ bd) {
  int idx = blockIdx.x * blockDim.x + threadIdx.x;     // L*H*S*S
  if (idx >= LL * HH * SS * SS) return;
  int r  = idx & 63;
  int s  = (idx >> 6) & 63;
  int lh = idx >> 12;                                  // l*4 + hh
  float acc = 0.f;
  for (int a = 0; a < AA; ++a)
    acc += bw[(lh * SS + s) * AA + a] * bw[(lh * SS + r) * AA + a];
  bd[idx] = acc;
}

// fac1[b,h,i,j] = bidir[1,h,i,j] * adj_pruned[b,h,i,j]
__global__ void k_fac1(const float* __restrict__ bd, const float* __restrict__ adj,
                       float* __restrict__ fac) {
  int idx = blockIdx.x * blockDim.x + threadIdx.x;     // B*H*4096
  if (idx >= BB * HH * SS * SS) return;
  int e  = idx & 4095;
  int hh = (idx >> 12) & 3;
  fac[idx] = bd[(size_t)(HH + hh) * 4096 + e] * adj[idx];
}

// ---------------- shared tile helpers ----------------

// builds hp (row-major, padded) and keyv TRANSPOSED (kvT[a][s], padded)
__device__ __forceinline__ void build_hp_kv(
    const _Float16* sh_h, _Float16* sh_hp, _Float16* sh_kvT,
    const float* sh_pe, const float* __restrict__ proj_w,
    const float* __restrict__ proj_b, const float* __restrict__ aw, int tid) {
  for (int i = tid; i < SS * 32; i += 128) {
    int s = i >> 5, a = i & 31;
    float acc = 0.f;
    if (a < AP) {
      acc = proj_b[a];
      for (int d = 0; d < DD; ++d)
        acc += (float)sh_h[s * HS + d] * proj_w[d * AP + a];
    }
    sh_hp[s * HPS + a] = (_Float16)acc;
  }
  for (int i = tid; i < SS * 32; i += 128) {
    int a = i & 31, s = i >> 5;
    float v = 0.f;
    if (a < AA) v = aw[s * AA + a];
    else if (a < AP) v = sh_pe[a - AA];
    sh_kvT[a * KVS + s] = (_Float16)v;
  }
}

// ---------------- layer apply (alpha -> final -> message passing), in place ----------------
// fac: per-element scale (bidir, or bidir*adj) with configurable batch stride (0 for layer 0)

__global__ __launch_bounds__(128) void k_tile(
    _Float16* __restrict__ hG, const float* __restrict__ peG,
    const float* __restrict__ attn_w, const float* __restrict__ proj_w,
    const float* __restrict__ proj_b, const float* __restrict__ fac,
    int facBstride, int lay) {
  __shared__ _Float16 sh_h[SS * HS];
  __shared__ _Float16 sh_hp[SS * HPS];
  __shared__ _Float16 sh_kvT[32 * KVS];
  __shared__ _Float16 sh_fin[SS * FINS];
  __shared__ float    sh_pe[PP];

  int tid = threadIdx.x;
  int bid = blockIdx.x;                 // ((b*4+hh)*128 + t)
  int t  = bid & 127;
  int hh = (bid >> 7) & 3;
  int b  = bid >> 9;
  size_t tile = (size_t)bid * (SS * DD);

  for (int i = tid; i < SS * DD; i += 128)
    sh_h[(i >> 4) * HS + (i & 15)] = hG[tile + i];
  if (tid < PP) sh_pe[tid] = peG[(b * TT + t) * PP + tid];
  __syncthreads();

  const float* aw = attn_w + (size_t)(lay * HH + hh) * SS * AA;
  build_hp_kv(sh_h, sh_hp, sh_kvT, sh_pe, proj_w, proj_b, aw, tid);
  __syncthreads();

  int wave = tid >> 5, lane = tid & 31;
  const float* fc = fac + (size_t)b * facBstride + (size_t)hh * 4096;

  // alpha = relu(hp @ keyv^T); final = fac * alpha  (f16 into sh_fin)
  {
    v16h a = load_a_frag(sh_hp, 16 * wave, HPS, lane);
#pragma unroll
    for (int j = 0; j < 4; ++j) {
      // B[k][n] = keyv[16j+n][k] = kvT[k][16j+n]  (contiguous per lane)
      v16h bf = load_b_frag(sh_kvT + 16 * j, 0, KVS, lane);
      v8f c = {0.f, 0.f, 0.f, 0.f, 0.f, 0.f, 0.f, 0.f};
      c = wmma_f16(a, bf, c);
      int mg0 = 16 * wave + ((lane >> 4) << 3);
      int ng  = 16 * j + (lane & 15);
#pragma unroll
      for (int v = 0; v < 8; ++v) {
        int mg = mg0 + v;
        float al = fmaxf(c[v], 0.f);
        sh_fin[mg * FINS + ng] = (_Float16)(al * fc[mg * SS + ng]);
      }
    }
  }
  __syncthreads();

  // h_new = relu(final @ h)   [64x64]@[64x16], K split 2x32
  {
    v16h a0 = load_a_frag(sh_fin,      16 * wave, FINS, lane);
    v16h a1 = load_a_frag(sh_fin + 32, 16 * wave, FINS, lane);
    v16h b0 = load_b_frag(sh_h, 0,  HS, lane);
    v16h b1 = load_b_frag(sh_h, 32, HS, lane);
    v8f c = {0.f, 0.f, 0.f, 0.f, 0.f, 0.f, 0.f, 0.f};
    c = wmma_f16(a0, b0, c);
    c = wmma_f16(a1, b1, c);
    int mg0 = 16 * wave + ((lane >> 4) << 3);
    int n   = lane & 15;
#pragma unroll
    for (int v = 0; v < 8; ++v)
      hG[tile + (size_t)(mg0 + v) * DD + n] = (_Float16)fmaxf(c[v], 0.f);
  }
}

// ---------------- layer-1 alpha sum over a chunk of t (deterministic partials) ----------------

__global__ __launch_bounds__(128) void k_alpha_sum(
    const _Float16* __restrict__ hG, const float* __restrict__ peG,
    const float* __restrict__ attn_w, const float* __restrict__ proj_w,
    const float* __restrict__ proj_b, float* __restrict__ adjp) {
  __shared__ _Float16 sh_h[SS * HS];
  __shared__ _Float16 sh_hp[SS * HPS];
  __shared__ _Float16 sh_kvT[32 * KVS];
  __shared__ float    accum[SS * SS];
  __shared__ float    sh_pe[PP];

  int tid = threadIdx.x;
  int bid = blockIdx.x;                 // B*H*8
  int tc = bid & 7;
  int hh = (bid >> 3) & 3;
  int b  = bid >> 5;

  for (int i = tid; i < SS * SS; i += 128) accum[i] = 0.f;
  const float* aw = attn_w + (size_t)(1 * HH + hh) * SS * AA;   // layer 1
  int wave = tid >> 5, lane = tid & 31;

  for (int tt = 0; tt < 16; ++tt) {
    int t = tc * 16 + tt;
    size_t tile = ((size_t)((b * HH + hh) * TT + t)) * (SS * DD);
    __syncthreads();
    for (int i = tid; i < SS * DD; i += 128)
      sh_h[(i >> 4) * HS + (i & 15)] = hG[tile + i];
    if (tid < PP) sh_pe[tid] = peG[(b * TT + t) * PP + tid];
    __syncthreads();
    build_hp_kv(sh_h, sh_hp, sh_kvT, sh_pe, proj_w, proj_b, aw, tid);
    __syncthreads();

    v16h a = load_a_frag(sh_hp, 16 * wave, HPS, lane);
#pragma unroll
    for (int j = 0; j < 4; ++j) {
      v16h bf = load_b_frag(sh_kvT + 16 * j, 0, KVS, lane);
      v8f c = {0.f, 0.f, 0.f, 0.f, 0.f, 0.f, 0.f, 0.f};
      c = wmma_f16(a, bf, c);
      int mg0 = 16 * wave + ((lane >> 4) << 3);
      int ng  = 16 * j + (lane & 15);
#pragma unroll
      for (int v = 0; v < 8; ++v) accum[(mg0 + v) * SS + ng] += fmaxf(c[v], 0.f);
    }
  }
  __syncthreads();
  float* out = adjp + (size_t)bid * (SS * SS);
  for (int i = tid; i < SS * SS; i += 128) out[i] = accum[i];
}

__global__ void k_adj_reduce(const float* __restrict__ adjp,
                             float* __restrict__ adj) {
  int idx = blockIdx.x * blockDim.x + threadIdx.x;     // B*H*4096
  if (idx >= BB * HH * SS * SS) return;
  int bh = idx >> 12, e = idx & 4095;
  float acc = 0.f;
  for (int tc = 0; tc < 8; ++tc) acc += adjp[((size_t)bh * 8 + tc) * 4096 + e];
  adj[idx] = acc;
}

// ---------------- prune: stable argsort-rank top-half keep ----------------

__global__ __launch_bounds__(256) void k_prune(float* __restrict__ adj,
                                               const float* __restrict__ msum) {
  __shared__ unsigned long long keys[4096];
  __shared__ unsigned char     keep[4096];
  int tid = threadIdx.x;
  int bid = blockIdx.x;                 // B*H
  int b = bid >> 2;
  float* ad = adj + (size_t)bid * 4096;
  const float* ms = msum + b * SS;

  for (int i = tid; i < 4096; i += 256) {
    int row = i >> 6;
    float v = ad[i] / ms[row];          // values are >= 0
    ad[i] = v;
    unsigned int bits = __float_as_uint(v);
    keys[i] = ((unsigned long long)bits << 12) | (unsigned)i;
    keep[i] = 0;
  }
  __syncthreads();
  // bitonic sort ascending (value, index) -> position == stable ascending rank
  for (int k = 2; k <= 4096; k <<= 1) {
    for (int j = k >> 1; j > 0; j >>= 1) {
      for (int i = tid; i < 4096; i += 256) {
        int ixj = i ^ j;
        if (ixj > i) {
          unsigned long long x = keys[i], y = keys[ixj];
          bool up = ((i & k) == 0);
          if ((x > y) == up) { keys[i] = y; keys[ixj] = x; }
        }
      }
      __syncthreads();
    }
  }
  for (int p = 2048 + tid; p < 4096; p += 256)
    keep[(int)(keys[p] & 0xFFFull)] = 1;
  __syncthreads();
  for (int i = tid; i < 4096; i += 256)
    if (!keep[i]) ad[i] = 0.f;
}

// ---------------- temporal attention + output ----------------

__global__ __launch_bounds__(128) void k_temporal(
    const _Float16* __restrict__ hG, const float* __restrict__ peG,
    const float* __restrict__ q_w, const float* __restrict__ q_b,
    const float* __restrict__ k_w, const float* __restrict__ k_b,
    const float* __restrict__ s_w, const float* __restrict__ s_b,
    const float* __restrict__ emb_w, const float* __restrict__ emb_b,
    float* __restrict__ outG) {
  __shared__ _Float16 Hc[TT * HCS];
  __shared__ float Qs[TT][16];
  __shared__ float score[TT];
  __shared__ float hbar[DP];
  __shared__ float kbar[16];
  __shared__ float out32[DP];
  __shared__ float ssum_s, qbdot_s, smax_s, sexp_s;

  int tid = threadIdx.x;
  int bid = blockIdx.x;                 // B*H*S
  int s  = bid & 63;
  int hh = (bid >> 6) & 3;
  int b  = bid >> 8;

  for (int i = tid; i < TT * DP; i += 128) {
    int t = i >> 5, c = i & 31;
    _Float16 v;
    if (c < DD)
      v = hG[(((size_t)(b * HH + hh) * TT + t) * SS + s) * DD + c];
    else
      v = (_Float16)peG[(b * TT + t) * PP + (c - DD)];
    Hc[t * HCS + c] = v;
  }
  __syncthreads();

  if (tid < DP) {
    float acc = 0.f;
    for (int t = 0; t < TT; ++t) acc += s_w[t] * (float)Hc[t * HCS + tid];
    hbar[tid] = acc;
  }
  if (tid == 32) { float ss = 0.f; for (int t = 0; t < TT; ++t) ss += s_w[t]; ssum_s = ss; }
  __syncthreads();
  if (tid < 16) {
    float v = 0.f;
    if (tid < TAH) {
      v = k_b[tid] * ssum_s;
      for (int c = 0; c < DP; ++c) v += hbar[c] * k_w[c * TAH + tid];
    }
    kbar[tid] = v;
  }
  __syncthreads();
  if (tid == 0) {
    float qd = 0.f;
    for (int n = 0; n < TAH; ++n) qd += q_b[n] * kbar[n];
    qbdot_s = qd;
  }

  // Q = Hc @ q_w via WMMA  (8 M-blocks over T, K=32, N=12 padded to 16)
  int wave = tid >> 5, lane = tid & 31;
#pragma unroll
  for (int mbi = 0; mbi < 2; ++mbi) {
    int mb = wave + 4 * mbi;
    v16h a = load_a_frag(Hc, 16 * mb, HCS, lane);
    v16h bf;
#pragma unroll
    for (int v = 0; v < 8; ++v) {
      int nq = lane & 15;
      int kq = ((v & 4) << 2) + ((lane >> 4) << 3) + ((v & 3) << 1);
      bf[2 * v]     = (_Float16)(nq < TAH ? q_w[kq * TAH + nq] : 0.f);
      bf[2 * v + 1] = (_Float16)(nq < TAH ? q_w[(kq + 1) * TAH + nq] : 0.f);
    }
    v8f c = {0.f, 0.f, 0.f, 0.f, 0.f, 0.f, 0.f, 0.f};
    c = wmma_f16(a, bf, c);
    int mg0 = 16 * mb + ((lane >> 4) << 3);
    int n   = lane & 15;
#pragma unroll
    for (int v = 0; v < 8; ++v) Qs[mg0 + v][n] = c[v];
  }
  __syncthreads();

  {
    float sc = qbdot_s;
    for (int n = 0; n < TAH; ++n) sc += Qs[tid][n] * kbar[n];
    score[tid] = sc * 0.17677669529663687f + s_b[0];   // 1/sqrt(32)
  }
  __syncthreads();
  if (tid == 0) {
    float mx = -3.0e38f;
    for (int t = 0; t < TT; ++t) mx = fmaxf(mx, score[t]);
    smax_s = mx;
  }
  __syncthreads();
  score[tid] = expf(score[tid] - smax_s);
  __syncthreads();
  if (tid == 0) {
    float ss = 0.f;
    for (int t = 0; t < TT; ++t) ss += score[t];
    sexp_s = ss;
  }
  __syncthreads();
  score[tid] /= sexp_s;                 // beta
  __syncthreads();
  if (tid < DP) {
    float acc = 0.f;
    for (int t = 0; t < TT; ++t) acc += score[t] * (float)Hc[t * HCS + tid];
    out32[tid] = acc;
  }
  __syncthreads();
  if (tid < OUTH) {
    float o = emb_b[tid];
    for (int c = 0; c < DP; ++c) o += out32[c] * emb_w[c * OUTH + tid];
    outG[((size_t)(b * SS + s)) * 64 + hh * OUTH + tid] = o;
  }
}

// ---------------- host ----------------

extern "C" void kernel_launch(void* const* d_in, const int* in_sizes, int n_in,
                              void* d_out, int out_size, void* d_ws, size_t ws_size,
                              hipStream_t stream) {
  (void)in_sizes; (void)n_in; (void)out_size; (void)ws_size;
  const float* x       = (const float*)d_in[0];
  const float* times   = (const float*)d_in[1];
  const float* mask    = (const float*)d_in[2];
  const float* obs_w   = (const float*)d_in[3];
  const float* attn_w  = (const float*)d_in[4];
  const float* proj_w  = (const float*)d_in[5];
  const float* proj_b  = (const float*)d_in[6];
  const float* bidir_w = (const float*)d_in[7];
  const float* q_w     = (const float*)d_in[8];
  const float* q_b     = (const float*)d_in[9];
  const float* k_w     = (const float*)d_in[10];
  const float* k_b     = (const float*)d_in[11];
  const float* s_w     = (const float*)d_in[12];
  const float* s_b     = (const float*)d_in[13];
  const float* emb_w   = (const float*)d_in[14];
  const float* emb_b   = (const float*)d_in[15];
  float* outG = (float*)d_out;

  char* ws = (char*)d_ws;
  const size_t SZ_H    = (size_t)BB * HH * TT * SS * DD * sizeof(_Float16); // 32 MB
  const size_t SZ_PE   = (size_t)BB * TT * PP * sizeof(float);
  const size_t SZ_MS   = (size_t)BB * SS * sizeof(float);
  const size_t SZ_BD   = (size_t)LL * HH * SS * SS * sizeof(float);
  const size_t SZ_ADJ  = (size_t)BB * HH * SS * SS * sizeof(float);
  const size_t SZ_ADJP = 8 * SZ_ADJ;
  _Float16* hG   = (_Float16*)(ws);
  float*    peG  = (float*)(ws + SZ_H);
  float*    msum = (float*)(ws + SZ_H + SZ_PE);
  float*    bd   = (float*)(ws + SZ_H + SZ_PE + SZ_MS);
  float*    adj  = (float*)(ws + SZ_H + SZ_PE + SZ_MS + SZ_BD);
  float*    adjp = (float*)(ws + SZ_H + SZ_PE + SZ_MS + SZ_BD + SZ_ADJ);
  float*    fac1 = (float*)(ws + SZ_H + SZ_PE + SZ_MS + SZ_BD + SZ_ADJ + SZ_ADJP);

  // precompute
  k_pe   <<<(BB * TT * 8 + 255) / 256, 256, 0, stream>>>(times, peG);
  k_h0   <<<(BB * HH * TT * SS * DD + 255) / 256, 256, 0, stream>>>(x, mask, obs_w, hG);
  k_msum <<<(BB * SS + 255) / 256, 256, 0, stream>>>(mask, msum);
  k_bidir<<<(LL * HH * SS * SS + 255) / 256, 256, 0, stream>>>(bidir_w, bd);

  // layer 0: fac = bidir[0,h]  (batch stride 0)
  k_tile<<<BB * HH * TT, 128, 0, stream>>>(hG, peG, attn_w, proj_w, proj_b,
                                           bd, 0, 0);
  // layer 1: pass A (alpha sums -> adj), prune, fuse fac, pass B (apply)
  k_alpha_sum <<<BB * HH * 8, 128, 0, stream>>>(hG, peG, attn_w, proj_w, proj_b, adjp);
  k_adj_reduce<<<(BB * HH * SS * SS + 255) / 256, 256, 0, stream>>>(adjp, adj);
  k_prune     <<<BB * HH, 256, 0, stream>>>(adj, msum);
  k_fac1      <<<(BB * HH * SS * SS + 255) / 256, 256, 0, stream>>>(bd, adj, fac1);
  k_tile      <<<BB * HH * TT, 128, 0, stream>>>(hG, peG, attn_w, proj_w, proj_b,
                                                 fac1, HH * SS * SS, 1);
  // temporal attention + output projection
  k_temporal<<<BB * HH * SS, 128, 0, stream>>>(hG, peG, q_w, q_b, k_w, k_b,
                                               s_w, s_b, emb_w, emb_b, outG);
}
